// CEABlock_Enhancement_52853867544591
// MI455X (gfx1250) — matrix operand order repaired and checked by reading, not compiled, not run on hardware
//
#include <hip/hip_runtime.h>
#include <math.h>

// ---------------------------------------------------------------------------
// Types for WMMA
// ---------------------------------------------------------------------------
typedef __attribute__((ext_vector_type(16))) __bf16 v16bf;
typedef __attribute__((ext_vector_type(8)))  float  v8f;
typedef __attribute__((ext_vector_type(4)))  __bf16 v4bf;
using bf16 = __bf16;

// ---------------------------------------------------------------------------
// Problem constants
// ---------------------------------------------------------------------------
constexpr int Bq  = 32;
constexpr int ZLq = 64;
constexpr int SLq = 256;
constexpr int Tq  = 320;   // ZL + SL
constexpr int TCq = 384;   // ZL + T
constexpr int Dq  = 768;
constexpr int Hq  = 12;
constexpr int HDq = 64;
constexpr int HIDq = 3072;

// ---------------------------------------------------------------------------
// Workspace layout (bytes)
// ---------------------------------------------------------------------------
constexpr size_t A256(size_t x){ return (x + 255) & ~(size_t)255; }

constexpr size_t OFF_WQKV   = 0;
constexpr size_t OFF_WPROJ  = OFF_WQKV   + A256((size_t)3*Dq*Dq*2);
constexpr size_t OFF_WFC1   = OFF_WPROJ  + A256((size_t)Dq*Dq*2);
constexpr size_t OFF_WFC2   = OFF_WFC1   + A256((size_t)HIDq*Dq*2);
constexpr size_t OFF_WAL    = OFF_WFC2   + A256((size_t)Dq*HIDq*2);
constexpr size_t OFF_WAL2   = OFF_WAL    + A256((size_t)Dq*2*Dq*2);
constexpr size_t OFF_WFUS   = OFF_WAL2   + A256((size_t)Dq*Dq*2);
constexpr size_t OFF_ZCAT   = OFF_WFUS   + A256((size_t)Dq*Dq*2);
constexpr size_t OFF_ZF32   = OFF_ZCAT   + A256((size_t)Bq*ZLq*2*Dq*2);
constexpr size_t OFF_ZFBF   = OFF_ZF32   + A256((size_t)Bq*ZLq*Dq*4);
constexpr size_t OFF_NSBF   = OFF_ZFBF   + A256((size_t)Bq*ZLq*Dq*2);
constexpr size_t OFF_NSIBF  = OFF_NSBF   + A256((size_t)Bq*SLq*Dq*2);
constexpr size_t OFF_AX     = OFF_NSIBF  + A256((size_t)Bq*SLq*Dq*2);
constexpr size_t OFF_AXI    = OFF_AX     + A256((size_t)Bq*ZLq*SLq*4);
constexpr size_t OFF_LNXC   = OFF_AXI    + A256((size_t)Bq*ZLq*SLq*4);
constexpr size_t OFF_LNXIC  = OFF_LNXC   + A256((size_t)Bq*TCq*Dq*2);
constexpr size_t OFF_QKV    = OFF_LNXIC  + A256((size_t)Bq*TCq*Dq*2);
constexpr size_t OFF_ATTNP  = OFF_QKV    + A256((size_t)Bq*TCq*3*Dq*2);
constexpr size_t OFF_HEADC  = OFF_ATTNP  + A256((size_t)Bq*Hq*TCq*TCq*2);
constexpr size_t OFF_XATTN  = OFF_HEADC  + A256((size_t)Bq*TCq*Dq*2);
constexpr size_t OFF_XIATTN = OFF_XATTN  + A256((size_t)Bq*TCq*Dq*4);
constexpr size_t OFF_VVBF   = OFF_XIATTN + A256((size_t)Bq*TCq*Dq*4);
constexpr size_t OFF_VF     = OFF_VVBF   + A256((size_t)Bq*SLq*Dq*2);
constexpr size_t OFF_LNVFC  = OFF_VF     + A256((size_t)Bq*SLq*Dq*4);
constexpr size_t OFF_ADAP2  = OFF_LNVFC  + A256((size_t)Bq*Tq*Dq*2);
constexpr size_t OFF_LNMID  = OFF_ADAP2  + A256((size_t)Bq*Tq*Dq*4);
constexpr size_t OFF_HBF    = OFF_LNMID  + A256((size_t)Bq*Tq*Dq*2);
constexpr size_t OFF_MLPT   = OFF_HBF    + A256((size_t)Bq*Tq*HIDq*2);
constexpr size_t WS_NEED    = OFF_MLPT   + A256((size_t)Bq*Tq*Dq*4);

// d_out element offsets (flat f32, reference return order)
constexpr long long O_X     = 0;
constexpr long long O_GIT   = O_X     + (long long)Bq*Tq*Dq;
constexpr long long O_GIS   = O_GIT   + (long long)Bq*ZLq;
constexpr long long O_ATTN  = O_GIS   + (long long)Bq*SLq;
constexpr long long O_XI    = O_ATTN  + (long long)Bq*Hq*TCq*TCq;
constexpr long long O_GITI  = O_XI    + (long long)Bq*Tq*Dq;
constexpr long long O_GISI  = O_GITI  + (long long)Bq*ZLq;
constexpr long long O_IATTN = O_GISI  + (long long)Bq*SLq;
constexpr long long O_ALPHA = O_IATTN + (long long)Bq*Hq*TCq*TCq;
constexpr long long O_UM    = O_ALPHA + (long long)Bq*SLq;
constexpr long long O_U     = O_UM    + (long long)Bq*SLq;
constexpr long long O_UI    = O_U     + (long long)Bq*SLq;

// ---------------------------------------------------------------------------
// Async memory->LDS staging helpers (CDNA5 GLOBAL_LOAD_ASYNC_TO_LDS_B128,
// tracked by ASYNCcnt; see CDNA5 ISA)
// ---------------------------------------------------------------------------
__device__ __forceinline__ void async_b128(const bf16* g, bf16* lds) {
    const unsigned long long ga = (unsigned long long)(const void*)g;
    const unsigned la = (unsigned)(size_t)lds;   // low 32 bits = LDS byte addr
    asm volatile("global_load_async_to_lds_b128 %0, %1, off"
                 :: "v"(la), "v"(ga) : "memory");
}

// Issue one K-slab of tile loads: 2 A-chunks + 2 B-chunks per thread (=4 async
// instructions per wave iteration -> wave ASYNCcnt += 4 per slab).
__device__ __forceinline__ void stage_slab(
    const bf16* __restrict__ A, const bf16* __restrict__ Bm,
    int rowBase, int colBase, int lda, int ldb, int k0, bool bT,
    bf16* as, bf16* bs, int tid)
{
    #pragma unroll
    for (int c = tid; c < 256; c += 128) {
        const int m  = c >> 2;
        const int ch = (c & 3) * 8;
        async_b128(A + (long long)(rowBase + m) * lda + k0 + ch, &as[m * 40 + ch]);
    }
    if (bT) {
        // B is NxK: stage n-major (fragments become contiguous ds_load_b128)
        #pragma unroll
        for (int c = tid; c < 256; c += 128) {
            const int n  = c >> 2;
            const int ch = (c & 3) * 8;
            async_b128(Bm + (long long)(colBase + n) * ldb + k0 + ch, &bs[n * 40 + ch]);
        }
    } else {
        // B is KxN: stage k-major
        #pragma unroll
        for (int c = tid; c < 256; c += 128) {
            const int kk = c >> 3;
            const int ch = (c & 7) * 8;
            async_b128(Bm + (long long)(k0 + kk) * ldb + colBase + ch, &bs[kk * 72 + ch]);
        }
    }
}

// ---------------------------------------------------------------------------
// Generic batched WMMA-bf16 GEMM: C = scale*(A @ B[^T]) (+bias) (+GELU)
//   flags bit0: B stored N x K row-major (compute A @ B^T)
//   flags bit1: add bias[n]
//   flags bit2: store bf16 (else f32)
//   flags bit3: exact GELU epilogue
// Block = 128 threads (4 waves); C tile 64x64, K slab 32, double-buffered
// async global->LDS staging. All M%64==0, N%64==0, K%32==0; all lda/ldb and
// batch strides are multiples of 8 elements, so 16B accesses are aligned.
// ---------------------------------------------------------------------------
__global__ __launch_bounds__(128) void gemm_bf16_wmma(
    const bf16* __restrict__ A, const bf16* __restrict__ Bm,
    const float* __restrict__ bias, void* __restrict__ Cout,
    int K, int lda, int ldb, int ldc,
    long long sAo, long long sAi, long long sBo, long long sBi,
    long long sCo, long long sCi, int batchInner,
    int flags, float scale)
{
    __shared__ bf16 As [2][64 * 40];   // [m][k], pitch 40 elem (80B)
    __shared__ bf16 Bsh[2][64 * 40];   // NT: [n][k] pitch 40; NN: [k][n] pitch 72

    const int bz = blockIdx.z;
    const int bo = bz / batchInner, bi = bz % batchInner;
    A  += bo * sAo + bi * sAi;
    Bm += bo * sBo + bi * sBi;
    const long long cbase = bo * sCo + bi * sCi;

    const int rowBase = blockIdx.y * 64;
    const int colBase = blockIdx.x * 64;
    const int tid  = threadIdx.x;
    const int lane = tid & 31;
    const int wave = tid >> 5;
    const bool bT  = (flags & 1) != 0;

    v8f acc[4];
    #pragma unroll
    for (int i = 0; i < 4; ++i) acc[i] = (v8f){0,0,0,0,0,0,0,0};

    const int half = lane >> 4;   // which 16-lane half
    const int nl   = lane & 15;
    const int nk   = K >> 5;      // number of 32-wide K slabs

    // prologue: stage slab 0 into buffer 0
    stage_slab(A, Bm, rowBase, colBase, lda, ldb, 0, bT, As[0], Bsh[0], tid);

    for (int kb = 0; kb < nk; ++kb) {
        const int cur = kb & 1;
        const bool more = (kb + 1) < nk;
        if (more) {
            // prefetch next slab into the other buffer (read last at kb-1)
            stage_slab(A, Bm, rowBase, colBase, lda, ldb, (kb + 1) * 32, bT,
                       As[cur ^ 1], Bsh[cur ^ 1], tid);
            asm volatile("s_wait_asynccnt 0x4" ::: "memory");  // slab kb done
        } else {
            asm volatile("s_wait_asynccnt 0x0" ::: "memory");  // all done
        }
        __syncthreads();   // every wave waited on its own async loads

        // ---- B fragment (K=32 x N=16 slab, this wave's 16 columns) ----
        // lanes 0-15: K 0..15 ; lanes 16-31: K 16..31 (2 K per VGPR)
        v16bf bfrag;
        if (bT) {
            const bf16* bp = &Bsh[cur][(wave * 16 + nl) * 40 + half * 16];
            #pragma unroll
            for (int j = 0; j < 16; ++j) bfrag[j] = bp[j];          // 2x ds_load_b128
        } else {
            #pragma unroll
            for (int j = 0; j < 16; ++j)
                bfrag[j] = Bsh[cur][(half * 16 + j) * 72 + wave * 16 + nl];
        }

        #pragma unroll
        for (int mt = 0; mt < 4; ++mt) {
            // ---- A fragment: contiguous 16-elem run, matching B ----
            // k = half*16 + 0..15
            const bf16* ap = &As[cur][(mt * 16 + nl) * 40 + half * 16];
            v16bf afrag;
            #pragma unroll
            for (int j = 0; j < 8; ++j) afrag[j]     = ap[j];       // ds_load_b128
            #pragma unroll
            for (int j = 0; j < 8; ++j) afrag[8 + j] = ap[8 + j];   // ds_load_b128
            acc[mt] = __builtin_amdgcn_wmma_f32_16x16x32_bf16(
                false, afrag, false, bfrag, (short)0, acc[mt], false, false);
        }
        __syncthreads();   // compute done before this buffer is refilled
    }

    // ---- epilogue ----
    const int col = colBase + wave * 16 + nl;
    const float bv = (flags & 2) ? bias[col] : 0.0f;
    #pragma unroll
    for (int mt = 0; mt < 4; ++mt) {
        #pragma unroll
        for (int r = 0; r < 8; ++r) {
            const int row = rowBase + mt * 16 + half * 8 + r;
            float v = acc[mt][r] * scale + bv;
            if (flags & 8) v = 0.5f * v * (1.0f + erff(v * 0.70710678118654752f));
            const long long off = cbase + (long long)row * ldc + col;
            if (flags & 4) ((bf16*)Cout)[off] = (bf16)v;
            else           ((float*)Cout)[off] = v;
        }
    }
}

// ---------------------------------------------------------------------------
// Elementwise / reduction kernels
// ---------------------------------------------------------------------------
// n must be a multiple of 4 (true for all uses here)
__global__ void cvt_f2bf(const float* __restrict__ in, bf16* __restrict__ out, long long n4) {
    long long i = (long long)blockIdx.x * 256 + threadIdx.x;
    if (i < n4) {
        const float4 f = ((const float4*)in)[i];
        v4bf o; o[0] = (bf16)f.x; o[1] = (bf16)f.y; o[2] = (bf16)f.z; o[3] = (bf16)f.w;
        ((v4bf*)out)[i] = o;
    }
}

__global__ void cvt_i2f(const int* __restrict__ in, float* __restrict__ out, long long n) {
    long long i = (long long)blockIdx.x * 256 + threadIdx.x;
    if (i < n) out[i] = (float)in[i];
}

// zcat[b,t,:768]=x[b,t], [768:]=xi[b,t]  (t < ZL), 4 elems/thread
__global__ void build_zcat(const float* __restrict__ x, const float* __restrict__ xi,
                           bf16* __restrict__ out) {
    long long q = (long long)blockIdx.x * 256 + threadIdx.x;  // index of 4-elem group
    const long long N4 = (long long)Bq * ZLq * 2 * Dq / 4;
    if (q >= N4) return;
    const long long i  = q * 4;
    const long long c  = i % (2 * Dq);
    const long long bt = i / (2 * Dq);
    const long long b  = bt / ZLq, t = bt % ZLq;
    const float4 f = (c < Dq)
        ? *(const float4*)(x  + (b * Tq + t) * Dq + c)
        : *(const float4*)(xi + (b * Tq + t) * Dq + (c - Dq));
    v4bf o; o[0] = (bf16)f.x; o[1] = (bf16)f.y; o[2] = (bf16)f.z; o[3] = (bf16)f.w;
    ((v4bf*)out)[q] = o;
}

// LayerNorm over 768, row = b*rowsPerBatch + t; source split between two tensors.
__global__ __launch_bounds__(256) void ln_rows(
    const float* __restrict__ src0, long long s0b,
    const float* __restrict__ src1, long long s1b,
    int split, int rowsPerBatch,
    const float* __restrict__ gamma, const float* __restrict__ beta,
    bf16* __restrict__ out_bf)
{
    __shared__ float red[256];
    const int row = blockIdx.x;
    const int b = row / rowsPerBatch, t = row % rowsPerBatch;
    const float* src = (t < split) ? src0 + (long long)b * s0b + (long long)t * Dq
                                   : src1 + (long long)b * s1b + (long long)(t - split) * Dq;
    const int tid = threadIdx.x;
    float v[3], s = 0.0f;
    #pragma unroll
    for (int j = 0; j < 3; ++j) { v[j] = src[tid + j * 256]; s += v[j]; }
    red[tid] = s; __syncthreads();
    for (int off = 128; off; off >>= 1) { if (tid < off) red[tid] += red[tid + off]; __syncthreads(); }
    const float mean = red[0] * (1.0f / Dq); __syncthreads();
    float sq = 0.0f;
    #pragma unroll
    for (int j = 0; j < 3; ++j) { const float d = v[j] - mean; sq += d * d; }
    red[tid] = sq; __syncthreads();
    for (int off = 128; off; off >>= 1) { if (tid < off) red[tid] += red[tid + off]; __syncthreads(); }
    const float inv = rsqrtf(red[0] * (1.0f / Dq) + 1e-5f);
    bf16* o = out_bf + (long long)row * Dq;
    #pragma unroll
    for (int j = 0; j < 3; ++j) {
        const int c = tid + j * 256;
        o[c] = (bf16)((v[j] - mean) * inv * gamma[c] + beta[c]);
    }
}

// row softmax over 384, in place + bf16 mirror
__global__ __launch_bounds__(256) void softmax384(float* __restrict__ data,
                                                  bf16* __restrict__ out_bf) {
    __shared__ float red[256];
    const long long row = blockIdx.x;
    float* p = data + row * (long long)TCq;
    const int tid = threadIdx.x;
    const float v0 = p[tid];
    const float v1 = (tid < 128) ? p[256 + tid] : -INFINITY;
    red[tid] = fmaxf(v0, v1); __syncthreads();
    for (int off = 128; off; off >>= 1) { if (tid < off) red[tid] = fmaxf(red[tid], red[tid + off]); __syncthreads(); }
    const float m = red[0]; __syncthreads();
    const float e0 = __expf(v0 - m);
    const float e1 = (tid < 128) ? __expf(v1 - m) : 0.0f;
    red[tid] = e0 + e1; __syncthreads();
    for (int off = 128; off; off >>= 1) { if (tid < off) red[tid] += red[tid + off]; __syncthreads(); }
    const float inv = 1.0f / red[0];
    const float r0 = e0 * inv;
    p[tid] = r0; out_bf[row * TCq + tid] = (bf16)r0;
    if (tid < 128) {
        const float r1 = e1 * inv;
        p[256 + tid] = r1; out_bf[row * TCq + 256 + tid] = (bf16)r1;
    }
}

// fusion: per (b,s) softmax over t=0..63 of attn_x/attn_xi, entropies, alpha
__global__ __launch_bounds__(32) void fusion_stats(
    const float* __restrict__ ax, const float* __restrict__ axi,
    float* __restrict__ u_o, float* __restrict__ ui_o,
    float* __restrict__ um_o, float* __restrict__ alpha_o)
{
    __shared__ float red[32];
    const int idx = blockIdx.x;            // b*SL + s
    const int b = idx >> 8, s = idx & 255;
    const int lane = threadIdx.x;
    const float* bases[2] = { ax + (long long)b * ZLq * SLq + s,
                              axi + (long long)b * ZLq * SLq + s };
    float u[2];
    for (int br = 0; br < 2; ++br) {
        const float a0 = bases[br][(long long)lane * SLq];
        const float a1 = bases[br][(long long)(lane + 32) * SLq];
        red[lane] = fmaxf(a0, a1); __syncthreads();
        for (int off = 16; off; off >>= 1) { if (lane < off) red[lane] = fmaxf(red[lane], red[lane + off]); __syncthreads(); }
        const float m = red[0]; __syncthreads();
        const float e0 = __expf(a0 - m), e1 = __expf(a1 - m);
        red[lane] = e0 + e1; __syncthreads();
        for (int off = 16; off; off >>= 1) { if (lane < off) red[lane] += red[lane + off]; __syncthreads(); }
        const float Z = red[0]; __syncthreads();
        const float p0 = e0 / Z, p1 = e1 / Z;
        red[lane] = -(p0 * __logf(p0 + 1e-8f) + p1 * __logf(p1 + 1e-8f)); __syncthreads();
        for (int off = 16; off; off >>= 1) { if (lane < off) red[lane] += red[lane + off]; __syncthreads(); }
        u[br] = red[0]; __syncthreads();
    }
    if (lane == 0) {
        u_o[idx]  = u[0];
        ui_o[idx] = u[1];
        um_o[idx] = 0.5f * (u[0] + u[1]);
        alpha_o[idx] = 1.0f / (1.0f + __expf(-(u[1] - u[0])));
    }
}

// vv = alpha*ns + (1-alpha)*nsi  (bf16), 4 elems/thread
__global__ void make_vv(const bf16* __restrict__ ns, const bf16* __restrict__ nsi,
                        const float* __restrict__ alpha, bf16* __restrict__ vv) {
    long long q = (long long)blockIdx.x * 256 + threadIdx.x;
    const long long N4 = (long long)Bq * SLq * Dq / 4;
    if (q >= N4) return;
    const long long row = (q * 4) / Dq;
    const float a = alpha[row];
    const v4bf n0 = ((const v4bf*)ns)[q];
    const v4bf n1 = ((const v4bf*)nsi)[q];
    v4bf o;
    #pragma unroll
    for (int j = 0; j < 4; ++j)
        o[j] = (bf16)(a * (float)n0[j] + (1.0f - a) * (float)n1[j]);
    ((v4bf*)vv)[q] = o;
}

// x_mid = x + x_attn[:,64:384] + v_fc ; xi_mid = xi + xi_attn[:,64:384] + v_fc
__global__ void assemble_mid(const float* __restrict__ x, const float* __restrict__ xi,
                             const float* __restrict__ xattn, const float* __restrict__ xiattn,
                             const float* __restrict__ vf,
                             float* __restrict__ xmid, float* __restrict__ ximid) {
    long long i = (long long)blockIdx.x * 256 + threadIdx.x;
    const long long N = (long long)Bq * Tq * Dq;
    if (i >= N) return;
    const int c = (int)(i % Dq);
    const long long bt = i / Dq;
    const int b = (int)(bt / Tq), t = (int)(bt % Tq);
    const float vfc = (t < ZLq)
        ? xiattn[((long long)b * TCq + t) * Dq + c]
        : vf[((long long)b * SLq + (t - ZLq)) * Dq + c];
    const float xa  = xattn [((long long)b * TCq + t + ZLq) * Dq + c];
    const float xia = xiattn[((long long)b * TCq + t + ZLq) * Dq + c];
    xmid[i]  = x[i]  + xa  + vfc;
    ximid[i] = xi[i] + xia + vfc;
}

__global__ void add2_inplace(float* __restrict__ dst, const float* __restrict__ a,
                             const float* __restrict__ b, long long n) {
    long long i = (long long)blockIdx.x * 256 + threadIdx.x;
    if (i < n) dst[i] += a[i] + b[i];
}

// ---------------------------------------------------------------------------
// Host launcher
// ---------------------------------------------------------------------------
static inline void launch_gemm(hipStream_t s, const bf16* A, const bf16* B,
    const float* bias, void* C, int M, int N, int K, int lda, int ldb, int ldc,
    long long sAo, long long sAi, long long sBo, long long sBi,
    long long sCo, long long sCi, int batchOuter, int batchInner,
    int flags, float scale)
{
    dim3 grid(N / 64, M / 64, batchOuter * batchInner);
    gemm_bf16_wmma<<<grid, 128, 0, s>>>(A, B, bias, C, K, lda, ldb, ldc,
        sAo, sAi, sBo, sBi, sCo, sCi, batchInner, flags, scale);
}

static inline int g1d(long long n) { return (int)((n + 255) / 256); }

extern "C" void kernel_launch(void* const* d_in, const int* in_sizes, int n_in,
                              void* d_out, int out_size, void* d_ws, size_t ws_size,
                              hipStream_t stream) {
    (void)in_sizes; (void)n_in; (void)out_size; (void)ws_size;
    const float* x      = (const float*)d_in[0];
    const float* xi     = (const float*)d_in[1];
    const int*   git    = (const int*)d_in[2];
    const int*   giti   = (const int*)d_in[3];
    const int*   gis    = (const int*)d_in[4];
    const int*   gisi   = (const int*)d_in[5];
    const float* w_qkv  = (const float*)d_in[6];
    const float* b_qkv  = (const float*)d_in[7];
    const float* w_proj = (const float*)d_in[8];
    const float* b_proj = (const float*)d_in[9];
    const float* g1v    = (const float*)d_in[10];
    const float* bt1    = (const float*)d_in[11];
    const float* g2v    = (const float*)d_in[12];
    const float* bt2    = (const float*)d_in[13];
    const float* w_fc1  = (const float*)d_in[14];
    const float* b_fc1  = (const float*)d_in[15];
    const float* w_fc2  = (const float*)d_in[16];
    const float* b_fc2  = (const float*)d_in[17];
    const float* w_al   = (const float*)d_in[18];
    const float* b_al   = (const float*)d_in[19];
    const float* w_al2  = (const float*)d_in[20];
    const float* b_al2  = (const float*)d_in[21];
    const float* w_fus  = (const float*)d_in[22];
    const float* b_fus  = (const float*)d_in[23];

    char* ws = (char*)d_ws;
    bf16*  WQKV  = (bf16*)(ws + OFF_WQKV);
    bf16*  WPROJ = (bf16*)(ws + OFF_WPROJ);
    bf16*  WFC1  = (bf16*)(ws + OFF_WFC1);
    bf16*  WFC2  = (bf16*)(ws + OFF_WFC2);
    bf16*  WAL   = (bf16*)(ws + OFF_WAL);
    bf16*  WAL2  = (bf16*)(ws + OFF_WAL2);
    bf16*  WFUS  = (bf16*)(ws + OFF_WFUS);
    bf16*  ZCAT  = (bf16*)(ws + OFF_ZCAT);
    float* ZF32  = (float*)(ws + OFF_ZF32);
    bf16*  ZFBF  = (bf16*)(ws + OFF_ZFBF);
    bf16*  NSBF  = (bf16*)(ws + OFF_NSBF);
    bf16*  NSIBF = (bf16*)(ws + OFF_NSIBF);
    float* AX    = (float*)(ws + OFF_AX);
    float* AXI   = (float*)(ws + OFF_AXI);
    bf16*  LNXC  = (bf16*)(ws + OFF_LNXC);
    bf16*  LNXIC = (bf16*)(ws + OFF_LNXIC);
    bf16*  QKV   = (bf16*)(ws + OFF_QKV);
    bf16*  ATTNP = (bf16*)(ws + OFF_ATTNP);
    bf16*  HEADC = (bf16*)(ws + OFF_HEADC);
    float* XATTN = (float*)(ws + OFF_XATTN);
    float* XIATT = (float*)(ws + OFF_XIATTN);
    bf16*  VVBF  = (bf16*)(ws + OFF_VVBF);
    float* VF    = (float*)(ws + OFF_VF);
    bf16*  LNVFC = (bf16*)(ws + OFF_LNVFC);
    float* ADAP2 = (float*)(ws + OFF_ADAP2);
    bf16*  LNMID = (bf16*)(ws + OFF_LNMID);
    bf16*  HBF   = (bf16*)(ws + OFF_HBF);
    float* MLPT  = (float*)(ws + OFF_MLPT);
    float* out   = (float*)d_out;

    const float inv_sqrt_d  = 0.03608439182435161f;  // 768^-0.5
    const float inv_sqrt_hd = 0.125f;                // 64^-0.5

    // --- index passthrough outputs (int -> float) ---
    cvt_i2f<<<g1d(Bq*ZLq), 256, 0, stream>>>(git,  out + O_GIT,  Bq*ZLq);
    cvt_i2f<<<g1d(Bq*SLq), 256, 0, stream>>>(gis,  out + O_GIS,  Bq*SLq);
    cvt_i2f<<<g1d(Bq*ZLq), 256, 0, stream>>>(giti, out + O_GITI, Bq*ZLq);
    cvt_i2f<<<g1d(Bq*SLq), 256, 0, stream>>>(gisi, out + O_GISI, Bq*SLq);

    // --- bf16 weight copies (4 elems / thread) ---
    cvt_f2bf<<<g1d(3LL*Dq*Dq/4), 256, 0, stream>>>(w_qkv,  WQKV,  3LL*Dq*Dq/4);
    cvt_f2bf<<<g1d((long long)Dq*Dq/4), 256, 0, stream>>>(w_proj, WPROJ, (long long)Dq*Dq/4);
    cvt_f2bf<<<g1d((long long)HIDq*Dq/4), 256, 0, stream>>>(w_fc1, WFC1, (long long)HIDq*Dq/4);
    cvt_f2bf<<<g1d((long long)Dq*HIDq/4), 256, 0, stream>>>(w_fc2, WFC2, (long long)Dq*HIDq/4);
    cvt_f2bf<<<g1d(2LL*Dq*Dq/4), 256, 0, stream>>>(w_al,   WAL,   2LL*Dq*Dq/4);
    cvt_f2bf<<<g1d((long long)Dq*Dq/4), 256, 0, stream>>>(w_al2,  WAL2,  (long long)Dq*Dq/4);
    cvt_f2bf<<<g1d((long long)Dq*Dq/4), 256, 0, stream>>>(w_fus,  WFUS,  (long long)Dq*Dq/4);

    // --- z_f = concat(z, zi) @ w_al^T + b_al ---
    build_zcat<<<g1d((long long)Bq*ZLq*2*Dq/4), 256, 0, stream>>>(x, xi, ZCAT);
    launch_gemm(stream, ZCAT, WAL, b_al, ZF32, Bq*ZLq, Dq, 2*Dq,
                2*Dq, 2*Dq, Dq, 0,0, 0,0, 0,0, 1,1, /*NT|BIAS*/3, 1.0f);
    cvt_f2bf<<<g1d((long long)Bq*ZLq*Dq/4), 256, 0, stream>>>(ZF32, ZFBF, (long long)Bq*ZLq*Dq/4);

    // --- ns / nsi = LN(s), LN(si) with g1 ---
    ln_rows<<<Bq*SLq, 256, 0, stream>>>(nullptr, 0, x  + (long long)ZLq*Dq, (long long)Tq*Dq,
                                        0, SLq, g1v, bt1, NSBF);
    ln_rows<<<Bq*SLq, 256, 0, stream>>>(nullptr, 0, xi + (long long)ZLq*Dq, (long long)Tq*Dq,
                                        0, SLq, g1v, bt1, NSIBF);

    // --- attn_x / attn_xi = (z_f @ ns^T) * D^-0.5  (per-batch) ---
    launch_gemm(stream, ZFBF, NSBF, nullptr, AX, ZLq, SLq, Dq, Dq, Dq, SLq,
                (long long)ZLq*Dq, 0, (long long)SLq*Dq, 0, (long long)ZLq*SLq, 0,
                Bq, 1, /*NT*/1, inv_sqrt_d);
    launch_gemm(stream, ZFBF, NSIBF, nullptr, AXI, ZLq, SLq, Dq, Dq, Dq, SLq,
                (long long)ZLq*Dq, 0, (long long)SLq*Dq, 0, (long long)ZLq*SLq, 0,
                Bq, 1, 1, inv_sqrt_d);

    // --- fusion stats + vv + v_f ---
    fusion_stats<<<Bq*SLq, 32, 0, stream>>>(AX, AXI, out + O_U, out + O_UI,
                                            out + O_UM, out + O_ALPHA);
    make_vv<<<g1d((long long)Bq*SLq*Dq/4), 256, 0, stream>>>(NSBF, NSIBF, out + O_ALPHA, VVBF);
    launch_gemm(stream, VVBF, WFUS, b_fus, VF, Bq*SLq, Dq, Dq, Dq, Dq, Dq,
                0,0, 0,0, 0,0, 1,1, 3, 1.0f);

    // --- two MHA branches (sequential; reuse QKV/ATTNP/HEADC) ---
    for (int br = 0; br < 2; ++br) {
        const float* src = br ? xi : x;
        bf16* lnc = br ? LNXIC : LNXC;
        float* sc = out + (br ? O_IATTN : O_ATTN);
        float* xatt = br ? XIATT : XATTN;

        // LN(concat(z_f, src)) with g1
        ln_rows<<<Bq*TCq, 256, 0, stream>>>(ZF32, (long long)ZLq*Dq,
                                            src, (long long)Tq*Dq,
                                            ZLq, TCq, g1v, bt1, lnc);
        // qkv = ln @ w_qkv^T + b_qkv  -> bf16
        launch_gemm(stream, lnc, WQKV, b_qkv, QKV, Bq*TCq, 3*Dq, Dq,
                    Dq, Dq, 3*Dq, 0,0, 0,0, 0,0, 1,1, /*NT|BIAS|BF16*/7, 1.0f);
        // scores = q @ k^T * HD^-0.5 -> f32 into d_out attn slot
        launch_gemm(stream, QKV, QKV + Hq*HDq, nullptr, sc, TCq, TCq, HDq,
                    3*Dq, 3*Dq, TCq,
                    (long long)TCq*3*Dq, HDq, (long long)TCq*3*Dq, HDq,
                    (long long)Hq*TCq*TCq, (long long)TCq*TCq,
                    Bq, Hq, /*NT*/1, inv_sqrt_hd);
        // softmax rows (in place) + bf16 mirror
        softmax384<<<Bq*Hq*TCq, 256, 0, stream>>>(sc, ATTNP);
        // head_out = attn @ v  (NN), write head-concat bf16 layout [b, n, h*64+d]
        launch_gemm(stream, ATTNP, QKV + 2*Hq*HDq, nullptr, HEADC, TCq, HDq, TCq,
                    TCq, 3*Dq, Dq,
                    (long long)Hq*TCq*TCq, (long long)TCq*TCq,
                    (long long)TCq*3*Dq, HDq,
                    (long long)TCq*Dq, HDq,
                    Bq, Hq, /*BF16 out*/4, 1.0f);
        // proj
        launch_gemm(stream, HEADC, WPROJ, b_proj, xatt, Bq*TCq, Dq, Dq,
                    Dq, Dq, Dq, 0,0, 0,0, 0,0, 1,1, 3, 1.0f);
    }

    // --- residual assembly into d_out x / xi slots ---
    assemble_mid<<<g1d((long long)Bq*Tq*Dq), 256, 0, stream>>>(
        x, xi, XATTN, XIATT, VF, out + O_X, out + O_XI);

    // --- adap2 = LN(v_fc, g2) @ w_al2^T + b_al2 ---
    ln_rows<<<Bq*Tq, 256, 0, stream>>>(XIATT, (long long)TCq*Dq,
                                       VF, (long long)SLq*Dq,
                                       ZLq, Tq, g2v, bt2, LNVFC);
    launch_gemm(stream, LNVFC, WAL2, b_al2, ADAP2, Bq*Tq, Dq, Dq,
                Dq, Dq, Dq, 0,0, 0,0, 0,0, 1,1, 3, 1.0f);

    // --- MLP + final residual for each stream ---
    for (int br = 0; br < 2; ++br) {
        float* mid = out + (br ? O_XI : O_X);
        ln_rows<<<Bq*Tq, 256, 0, stream>>>(nullptr, 0, mid, (long long)Tq*Dq,
                                           0, Tq, g2v, bt2, LNMID);
        // h = gelu(ln @ w_fc1^T + b_fc1) -> bf16
        launch_gemm(stream, LNMID, WFC1, b_fc1, HBF, Bq*Tq, HIDq, Dq,
                    Dq, Dq, HIDq, 0,0, 0,0, 0,0, 1,1, /*NT|BIAS|BF16|GELU*/15, 1.0f);
        // mlp = h @ w_fc2^T + b_fc2 -> f32
        launch_gemm(stream, HBF, WFC2, b_fc2, MLPT, Bq*Tq, Dq, HIDq,
                    HIDq, HIDq, Dq, 0,0, 0,0, 0,0, 1,1, 3, 1.0f);
        add2_inplace<<<g1d((long long)Bq*Tq*Dq), 256, 0, stream>>>(
            mid, MLPT, ADAP2, (long long)Bq*Tq*Dq);
    }
}
